// MultiHeadAttention_76459007803543
// MI455X (gfx1250) — matrix operand order repaired and checked by reading, not compiled, not run on hardware
//
#include <hip/hip_runtime.h>
#include <hip/hip_bf16.h>

typedef _Float16 half_t;
typedef _Float16 v16h __attribute__((ext_vector_type(16)));
typedef float    v8f  __attribute__((ext_vector_type(8)));

#define D_MODEL 1024
#define N_HEADS 16
#define D_KH    64
#define BATCH   2
#define SEQ     2048
#define ROWS    (BATCH*SEQ)     // 4096
#define N_QKV   (3*D_MODEL)     // 3072

// ---------------- Tensor Data Mover support ----------------
#if defined(__has_builtin)
# if __has_builtin(__builtin_amdgcn_tensor_load_to_lds)
#  define HAVE_TDM 1
# endif
#endif
#ifndef HAVE_TDM
# define HAVE_TDM 0
#endif

#if HAVE_TDM
typedef unsigned int u32;
typedef u32 u32x4 __attribute__((ext_vector_type(4)));
typedef int  i32x4 __attribute__((ext_vector_type(4)));
typedef int  i32x8 __attribute__((ext_vector_type(8)));

// 2D f16 tile load, global -> LDS, per CDNA5 D# layout (ISA 08_async_tensor §8).
__device__ __forceinline__ void tdm_load_2d(void* lds_ptr, const half_t* gptr,
                                            u32 tile_w, u32 tile_h,
                                            u32 tensor_w, u32 tensor_h, u32 stride_e,
                                            u32 pad_int, u32 pad_amt, bool pad_en) {
  unsigned long long ga = (unsigned long long)(size_t)gptr;
  u32x4 g0;
  g0[0] = 1u;                                            // count=1, user mode
  g0[1] = (u32)(size_t)lds_ptr;                          // lds_addr (bytes)
  g0[2] = (u32)ga;                                       // global_addr[31:0]
  g0[3] = (u32)((ga >> 32) & 0x01FFFFFFu) | (2u << 30);  // addr[56:32] | type=2
  i32x8 g1;
  g1[0] = (int)((1u << 16) |                             // data_size = 2B
                (pad_en ? ((1u << 20) | (pad_int << 22) | (pad_amt << 25)) : 0u));
  g1[1] = (int)(tensor_w << 16);                         // tensor_dim0[15:0]
  g1[2] = (int)((tensor_w >> 16) | (tensor_h << 16));    // dim0[31:16] | dim1[15:0]
  g1[3] = (int)((tensor_h >> 16) | (tile_w << 16));      // dim1[31:16] | tile_dim0
  g1[4] = (int)tile_h;                                   // tile_dim1 (tile_dim2=0)
  g1[5] = (int)stride_e;                                 // tensor_dim0_stride[31:0]
  g1[6] = 0;
  g1[7] = 0;
  i32x4 z4 = {};
#if __clang_major__ >= 23
  i32x8 z8 = {};
  __builtin_amdgcn_tensor_load_to_lds(g0, g1, z4, z4, z8, 0);
#else
  __builtin_amdgcn_tensor_load_to_lds(g0, g1, z4, z4, 0);
#endif
}
#endif  // HAVE_TDM

__device__ __forceinline__ v8f wmma16(v16h a, v16h b, v8f c) {
  return __builtin_amdgcn_wmma_f32_16x16x32_f16(false, a, false, b, (short)0, c,
                                                false, false);
}

// ---- fragment loaders per CDNA5 ISA 7.12.2 layouts (wave32) ----
// A fragment 16x32 f16 from row-major [M][K] tile: contiguous pairs -> ds_load_b128 x2
__device__ __forceinline__ v16h load_a_frag(const half_t* t, int ld, int m0, int k0, int lane) {
  int m  = m0 + (lane & 15);
  int hh = lane >> 4;
  v16h a;
#pragma unroll
  for (int v = 0; v < 8; ++v) {
    int kb = ((v < 4) ? (2 * v) : (16 + 2 * (v - 4))) + hh * 8;
    a[2 * v]     = t[m * ld + k0 + kb];
    a[2 * v + 1] = t[m * ld + k0 + kb + 1];
  }
  return a;
}

// B fragment where logical B[k][n] = t[n][k] (k contiguous per lane -> ds_load_b128 x2)
__device__ __forceinline__ v16h load_bT_frag(const half_t* t, int ld, int n0, int k0, int lane) {
  int n  = n0 + (lane & 15);
  int hh = lane >> 4;
  v16h b;
#pragma unroll
  for (int e = 0; e < 16; ++e)
    b[e] = t[(size_t)n * ld + k0 + ((e < 8) ? e : (8 + e)) + hh * 8];
  return b;
}

// ---------------- stage 0: f32 -> f16 (vectorized x4) ----------------
__global__ void cvt_f32_f16(const float* __restrict__ s, half_t* __restrict__ d, int n4) {
  int i = blockIdx.x * 256 + threadIdx.x;
  if (i < n4) {
    float4 v = ((const float4*)s)[i];
    half_t* o = d + (size_t)i * 4;
    o[0] = (half_t)v.x; o[1] = (half_t)v.y; o[2] = (half_t)v.z; o[3] = (half_t)v.w;
  }
}

// ---------------- GEMM core used by stage 1 & 3 ----------------
// Block 128x128, 8 waves as 2(m) x 4(n). A (rows x K) TDM-loaded row-major [M][K] ld=40.
// B (K x cols) coop-loaded from global and stored TRANSPOSED [N][K] ld=40 so both
// fragment types are contiguous ds_load_b128 reads. Double-buffered, 1 barrier/iter.

#define GLDA 40
#define GLDB 40

// ---------------- stage 1: QKV GEMM (4096x1024 @ 1024x3072) ----------------
__global__ __launch_bounds__(256) void qkv_gemm(const half_t* __restrict__ X,
                                                const half_t* __restrict__ W,
                                                half_t* __restrict__ Qh,
                                                half_t* __restrict__ Kh,
                                                half_t* __restrict__ VhT) {
  __shared__ half_t At[2][128 * GLDA];
  __shared__ half_t Bt[2][128 * GLDB];   // transposed W tile: [n][k]

  int tid = threadIdx.x;
  int wave = tid >> 5, lane = tid & 31;
  int wm = wave >> 2, wn = wave & 3;
  int m0 = blockIdx.y * 128;
  int n0 = blockIdx.x * 128;

  v8f acc[4][2] = {};

  int brow = tid >> 3, bseg = tid & 7;   // B coop: k-row 0..31, 16-col segment

  // prologue: A tile 0 via TDM (wave0), B tile 0 coop-transposed
#if HAVE_TDM
  if (wave == 0)
    tdm_load_2d(&At[0][0], &X[(size_t)m0 * D_MODEL], 32, 128, D_MODEL, ROWS, D_MODEL, 3, 3, true);
#else
  {
    int ar = tid >> 1, as = (tid & 1) * 16;
    *(uint4*)&At[0][ar * GLDA + as] = *(const uint4*)&X[(size_t)(m0 + ar) * D_MODEL + as];
  }
#endif
  {
    uint4 wv = *(const uint4*)&W[(size_t)brow * N_QKV + n0 + bseg * 16];
    const half_t* wh = (const half_t*)&wv;
#pragma unroll
    for (int j = 0; j < 16; ++j)
      Bt[0][(bseg * 16 + j) * GLDB + brow] = wh[j];
  }
#if HAVE_TDM
  if (wave == 0) __builtin_amdgcn_s_wait_tensorcnt(0);
#endif
  __syncthreads();

  for (int kt = 0; kt < D_MODEL; kt += 32) {
    int cur = (kt >> 5) & 1, nxt = cur ^ 1;
    if (kt + 32 < D_MODEL) {   // pipeline next tiles into nxt
#if HAVE_TDM
      if (wave == 0)
        tdm_load_2d(&At[nxt][0], &X[(size_t)m0 * D_MODEL + kt + 32], 32, 128, D_MODEL, ROWS, D_MODEL, 3, 3, true);
#else
      int ar = tid >> 1, as = (tid & 1) * 16;
      *(uint4*)&At[nxt][ar * GLDA + as] = *(const uint4*)&X[(size_t)(m0 + ar) * D_MODEL + kt + 32 + as];
#endif
      uint4 wv = *(const uint4*)&W[(size_t)(kt + 32 + brow) * N_QKV + n0 + bseg * 16];
      const half_t* wh = (const half_t*)&wv;
#pragma unroll
      for (int j = 0; j < 16; ++j)
        Bt[nxt][(bseg * 16 + j) * GLDB + brow] = wh[j];
    }
    const half_t* Ac = &At[cur][0];
    const half_t* Bc = &Bt[cur][0];
    v16h bf[2];
#pragma unroll
    for (int nt = 0; nt < 2; ++nt)
      bf[nt] = load_bT_frag(Bc, GLDB, wn * 32 + nt * 16, 0, lane);
#pragma unroll
    for (int mt = 0; mt < 4; ++mt) {
      v16h af = load_a_frag(Ac, GLDA, wm * 64 + mt * 16, 0, lane);
#pragma unroll
      for (int nt = 0; nt < 2; ++nt)
        acc[mt][nt] = wmma16(af, bf[nt], acc[mt][nt]);
    }
#if HAVE_TDM
    if (wave == 0) __builtin_amdgcn_s_wait_tensorcnt(0);
#endif
    __syncthreads();
  }

  // epilogue: part uniform per block; V written TRANSPOSED [B,H,dk,T]
  int part = blockIdx.x >> 3;                       // n0 / 1024
  half_t* dst = (part == 0) ? Qh : ((part == 1) ? Kh : VhT);
  float scl   = (part == 0) ? 0.125f : 1.0f;        // fold 1/sqrt(dk) into Q
  size_t st_t = (part == 2) ? 1 : D_KH;             // stride of t
  size_t st_d = (part == 2) ? SEQ : 1;              // stride of d
  int ncol0   = n0 - part * D_MODEL;

  int hh = lane >> 4, ln = lane & 15;
#pragma unroll
  for (int mt = 0; mt < 4; ++mt)
#pragma unroll
    for (int nt = 0; nt < 2; ++nt)
#pragma unroll
      for (int r = 0; r < 8; ++r) {
        int row  = m0 + wm * 64 + mt * 16 + r + 8 * hh;     // b*T + t
        int col  = ncol0 + wn * 32 + nt * 16 + ln;          // 0..1023 within part
        int b    = row >> 11, t = row & (SEQ - 1);
        int head = col >> 6,  d = col & 63;
        size_t idx = (size_t)(b * N_HEADS + head) * (D_KH * SEQ) + (size_t)t * st_t + (size_t)d * st_d;
        dst[idx] = (half_t)(acc[mt][nt][r] * scl);
      }
}

// ---------------- stage 2: causal flash attention ----------------
__global__ __launch_bounds__(256) void attn_kernel(const half_t* __restrict__ Qh,
                                                   const half_t* __restrict__ Kh,
                                                   const half_t* __restrict__ VhT,
                                                   half_t* __restrict__ AOh) {
  __shared__ half_t Kt[2][32 * 72];     // K tile [t][d] row-major, ld=72 (TDM pad)
  __shared__ half_t Vt[2][64 * 40];     // V tile [d][t] (global V is transposed), ld=40
  __shared__ half_t Pt[8 * 16 * 40];    // per-wave 16x32 P staging, ld=40 (16B rows)

  int qb = blockIdx.x;
  int bh = blockIdx.y;
  int b  = bh >> 4, h = bh & 15;
  const half_t* Qp  = Qh  + (size_t)bh * SEQ * D_KH;
  const half_t* Kp  = Kh  + (size_t)bh * SEQ * D_KH;
  const half_t* VpT = VhT + (size_t)bh * D_KH * SEQ;

  int tid = threadIdx.x, wave = tid >> 5, lane = tid & 31;
  int hh = lane >> 4, ln = lane & 15;
  int qbase = qb * 128 + wave * 16;
  const float NEG = -__builtin_inff();

  // wave's Q strip (16x64) as two A fragments straight from global
  v16h qf[2];
#pragma unroll
  for (int kc = 0; kc < 2; ++kc)
#pragma unroll
    for (int v = 0; v < 8; ++v) {
      int kk = ((v < 4) ? (2 * v) : (16 + 2 * (v - 4))) + hh * 8 + kc * 32;
      const half_t* qrow = &Qp[(size_t)(qbase + ln) * D_KH];
      qf[kc][2 * v]     = qrow[kk];
      qf[kc][2 * v + 1] = qrow[kk + 1];
    }

  float mrow[8], lrow[8];
  v8f   oacc[4] = {};
#pragma unroll
  for (int r = 0; r < 8; ++r) { mrow[r] = NEG; lrow[r] = 0.f; }

  int kbN = (qb + 1) * 4;   // causal block bound

#if HAVE_TDM
  if (wave == 0) {
    tdm_load_2d(&Kt[0][0], Kp,  64, 32, 64,  SEQ,  64,  4, 3, true);   // [32 t][64 d]
    tdm_load_2d(&Vt[0][0], VpT, 32, 64, SEQ, D_KH, SEQ, 3, 3, true);   // [64 d][32 t]
    __builtin_amdgcn_s_wait_tensorcnt(0);
  }
  __syncthreads();
#else
  int kr = tid >> 3, ks = (tid & 7) * 8;   // K coop: 32 rows x 8 segs
  int vr = tid >> 2, vs = (tid & 3) * 8;   // V coop: 64 rows x 4 segs
#endif

  for (int kb = 0; kb < kbN; ++kb) {
    int cur = kb & 1;
#if HAVE_TDM
    int nxt = cur ^ 1;
    if (kb + 1 < kbN && wave == 0) {      // async prefetch of next K/V tiles
      tdm_load_2d(&Kt[nxt][0], Kp + (size_t)(kb + 1) * 32 * D_KH, 64, 32, 64,  SEQ,  64,  4, 3, true);
      tdm_load_2d(&Vt[nxt][0], VpT + (size_t)(kb + 1) * 32,       32, 64, SEQ, D_KH, SEQ, 3, 3, true);
    }
#else
    cur = 0;
    *(uint4*)&Kt[0][kr * 72 + ks] = *(const uint4*)&Kp[(size_t)(kb * 32 + kr) * D_KH + ks];
    *(uint4*)&Vt[0][vr * 40 + vs] = *(const uint4*)&VpT[(size_t)vr * SEQ + kb * 32 + vs];
    __syncthreads();
#endif
    const half_t* Kc = &Kt[cur][0];
    const half_t* Vc = &Vt[cur][0];

    // S = Q*K^T (16x32 strip)
    v8f s[2] = {};
#pragma unroll
    for (int nt = 0; nt < 2; ++nt)
#pragma unroll
      for (int kc = 0; kc < 2; ++kc) {
        v16h bf = load_bT_frag(Kc, 72, nt * 16, kc * 32, lane);
        s[nt] = wmma16(qf[kc], bf, s[nt]);
      }

    // streaming causal softmax
    float corr[8];
#pragma unroll
    for (int r = 0; r < 8; ++r) {
      int q  = qbase + r + 8 * hh;
      float v0 = s[0][r]; if (kb * 32 + ln > q)      v0 = NEG;
      float v1 = s[1][r]; if (kb * 32 + 16 + ln > q) v1 = NEG;
      float mx = fmaxf(v0, v1);
      mx = fmaxf(mx, __shfl_xor(mx, 1, 32));
      mx = fmaxf(mx, __shfl_xor(mx, 2, 32));
      mx = fmaxf(mx, __shfl_xor(mx, 4, 32));
      mx = fmaxf(mx, __shfl_xor(mx, 8, 32));
      float mnew = fmaxf(mrow[r], mx);
      float c  = (mrow[r] == NEG) ? 0.f : __expf(mrow[r] - mnew);
      float p0 = (v0 == NEG) ? 0.f : __expf(v0 - mnew);
      float p1 = (v1 == NEG) ? 0.f : __expf(v1 - mnew);
      float ps = p0 + p1;
      ps += __shfl_xor(ps, 1, 32);
      ps += __shfl_xor(ps, 2, 32);
      ps += __shfl_xor(ps, 4, 32);
      ps += __shfl_xor(ps, 8, 32);
      lrow[r] = lrow[r] * c + ps;
      mrow[r] = mnew;
      corr[r] = c;
      s[0][r] = p0; s[1][r] = p1;
    }
#pragma unroll
    for (int dt = 0; dt < 4; ++dt)
#pragma unroll
      for (int r = 0; r < 8; ++r) oacc[dt][r] *= corr[r];

    // restage P: C-layout -> LDS -> A-layout (wave-private; LDS in-order per wave)
    half_t* P = &Pt[wave * 16 * 40];
#pragma unroll
    for (int nt = 0; nt < 2; ++nt)
#pragma unroll
      for (int r = 0; r < 8; ++r)
        P[(r + 8 * hh) * 40 + nt * 16 + ln] = (half_t)s[nt][r];

    v16h pa = load_a_frag(P, 40, 0, 0, lane);
#pragma unroll
    for (int dt = 0; dt < 4; ++dt) {
      v16h bv = load_bT_frag(Vc, 40, dt * 16, 0, lane);   // B[k][d] = Vt[d][k]
      oacc[dt] = wmma16(pa, bv, oacc[dt]);
    }
#if HAVE_TDM
    if (wave == 0) __builtin_amdgcn_s_wait_tensorcnt(0);
#endif
    __syncthreads();
  }

  // normalize + write concatenated heads
#pragma unroll
  for (int r = 0; r < 8; ++r) lrow[r] = 1.f / lrow[r];
#pragma unroll
  for (int dt = 0; dt < 4; ++dt)
#pragma unroll
    for (int r = 0; r < 8; ++r) {
      int t   = qbase + r + 8 * hh;
      int col = h * D_KH + dt * 16 + ln;
      AOh[((size_t)(b * SEQ + t)) * D_MODEL + col] = (half_t)(oacc[dt][r] * lrow[r]);
    }
}

// ---------------- stage 3: output projection -> f32 ----------------
__global__ __launch_bounds__(256) void out_gemm(const half_t* __restrict__ A,
                                                const half_t* __restrict__ W,
                                                float* __restrict__ out) {
  __shared__ half_t At[2][128 * GLDA];
  __shared__ half_t Bt[2][128 * GLDB];

  int tid = threadIdx.x;
  int wave = tid >> 5, lane = tid & 31;
  int wm = wave >> 2, wn = wave & 3;
  int m0 = blockIdx.y * 128;
  int n0 = blockIdx.x * 128;

  v8f acc[4][2] = {};
  int brow = tid >> 3, bseg = tid & 7;

#if HAVE_TDM
  if (wave == 0)
    tdm_load_2d(&At[0][0], &A[(size_t)m0 * D_MODEL], 32, 128, D_MODEL, ROWS, D_MODEL, 3, 3, true);
#else
  {
    int ar = tid >> 1, as = (tid & 1) * 16;
    *(uint4*)&At[0][ar * GLDA + as] = *(const uint4*)&A[(size_t)(m0 + ar) * D_MODEL + as];
  }
#endif
  {
    uint4 wv = *(const uint4*)&W[(size_t)brow * D_MODEL + n0 + bseg * 16];
    const half_t* wh = (const half_t*)&wv;
#pragma unroll
    for (int j = 0; j < 16; ++j)
      Bt[0][(bseg * 16 + j) * GLDB + brow] = wh[j];
  }
#if HAVE_TDM
  if (wave == 0) __builtin_amdgcn_s_wait_tensorcnt(0);
#endif
  __syncthreads();

  for (int kt = 0; kt < D_MODEL; kt += 32) {
    int cur = (kt >> 5) & 1, nxt = cur ^ 1;
    if (kt + 32 < D_MODEL) {
#if HAVE_TDM
      if (wave == 0)
        tdm_load_2d(&At[nxt][0], &A[(size_t)m0 * D_MODEL + kt + 32], 32, 128, D_MODEL, ROWS, D_MODEL, 3, 3, true);
#else
      int ar = tid >> 1, as = (tid & 1) * 16;
      *(uint4*)&At[nxt][ar * GLDA + as] = *(const uint4*)&A[(size_t)(m0 + ar) * D_MODEL + kt + 32 + as];
#endif
      uint4 wv = *(const uint4*)&W[(size_t)(kt + 32 + brow) * D_MODEL + n0 + bseg * 16];
      const half_t* wh = (const half_t*)&wv;
#pragma unroll
      for (int j = 0; j < 16; ++j)
        Bt[nxt][(bseg * 16 + j) * GLDB + brow] = wh[j];
    }
    const half_t* Ac = &At[cur][0];
    const half_t* Bc = &Bt[cur][0];
    v16h bf[2];
#pragma unroll
    for (int nt = 0; nt < 2; ++nt)
      bf[nt] = load_bT_frag(Bc, GLDB, wn * 32 + nt * 16, 0, lane);
#pragma unroll
    for (int mt = 0; mt < 4; ++mt) {
      v16h af = load_a_frag(Ac, GLDA, wm * 64 + mt * 16, 0, lane);
#pragma unroll
      for (int nt = 0; nt < 2; ++nt)
        acc[mt][nt] = wmma16(af, bf[nt], acc[mt][nt]);
    }
#if HAVE_TDM
    if (wave == 0) __builtin_amdgcn_s_wait_tensorcnt(0);
#endif
    __syncthreads();
  }

  int hh = lane >> 4, ln = lane & 15;
#pragma unroll
  for (int mt = 0; mt < 4; ++mt)
#pragma unroll
    for (int nt = 0; nt < 2; ++nt)
#pragma unroll
      for (int r = 0; r < 8; ++r) {
        int row = m0 + wm * 64 + mt * 16 + r + 8 * hh;
        int col = n0 + wn * 32 + nt * 16 + ln;
        out[(size_t)row * D_MODEL + col] = acc[mt][nt][r];
      }
}

// ---------------- host ----------------
extern "C" void kernel_launch(void* const* d_in, const int* in_sizes, int n_in,
                              void* d_out, int out_size, void* d_ws, size_t ws_size,
                              hipStream_t stream) {
  (void)in_sizes; (void)n_in; (void)out_size; (void)ws_size;

  const float* x     = (const float*)d_in[0];
  const float* W_qkv = (const float*)d_in[1];
  const float* W_o   = (const float*)d_in[2];
  float*       out   = (float*)d_out;

  char* p = (char*)d_ws;
  half_t* xh    = (half_t*)p;  p += (size_t)ROWS * D_MODEL * sizeof(half_t);
  half_t* wqkvh = (half_t*)p;  p += (size_t)D_MODEL * N_QKV * sizeof(half_t);
  half_t* woh   = (half_t*)p;  p += (size_t)D_MODEL * D_MODEL * sizeof(half_t);
  half_t* qh    = (half_t*)p;  p += (size_t)ROWS * D_MODEL * sizeof(half_t);
  half_t* kh    = (half_t*)p;  p += (size_t)ROWS * D_MODEL * sizeof(half_t);
  half_t* vht   = (half_t*)p;  p += (size_t)ROWS * D_MODEL * sizeof(half_t);  // [B,H,dk,T]
  half_t* aoh   = (half_t*)p;  p += (size_t)ROWS * D_MODEL * sizeof(half_t);

  int nx = ROWS * D_MODEL / 4, nq = D_MODEL * N_QKV / 4, no = D_MODEL * D_MODEL / 4;
  cvt_f32_f16<<<(nx + 255) / 256, 256, 0, stream>>>(x, xh, nx);
  cvt_f32_f16<<<(nq + 255) / 256, 256, 0, stream>>>(W_qkv, wqkvh, nq);
  cvt_f32_f16<<<(no + 255) / 256, 256, 0, stream>>>(W_o, woh, no);

  qkv_gemm<<<dim3(N_QKV / 128, ROWS / 128), 256, 0, stream>>>(xh, wqkvh, qh, kh, vht);
  attn_kernel<<<dim3(SEQ / 128, BATCH * N_HEADS), 256, 0, stream>>>(qh, kh, vht, aoh);
  out_gemm<<<dim3(D_MODEL / 128, ROWS / 128), 256, 0, stream>>>(aoh, woh, out);
}
